// SparseTransformerBlock_3496103379664
// MI455X (gfx1250) — compile-verified, matrix operand order checked by reading, not run on hardware
//
#include <hip/hip_runtime.h>
#include <hip/hip_bf16.h>

typedef __attribute__((ext_vector_type(16))) _Float16 v16h;
typedef __attribute__((ext_vector_type(8)))  _Float16 v8h;
typedef __attribute__((ext_vector_type(8)))  float    v8f;

union fragu { v16h v; v8h h[2]; };

__device__ __forceinline__ v8f wmma_f16(v16h a, v16h b, v8f c) {
    // D = A(16x32 f16) * B(32x16 f16) + C(16x16 f32)
    return __builtin_amdgcn_wmma_f32_16x16x32_f16(
        /*neg_a=*/false, a, /*neg_b=*/false, b,
        /*c_mod=*/(short)0, c, /*reuse_a=*/false, /*reuse_b=*/false);
}

#if __has_builtin(__builtin_amdgcn_tensor_load_to_lds) && __has_builtin(__builtin_amdgcn_s_wait_tensorcnt)
#define HAVE_TDM 1
typedef __attribute__((ext_vector_type(4))) unsigned int v4u;
typedef __attribute__((ext_vector_type(8))) int v8i;
typedef __attribute__((ext_vector_type(4))) int v4i;

// Tensor-DMA a [rows x 32]-f16 tile (row stride K elems in memory) into LDS at
// byte offset lds_off, padding each 64B row by 16B -> LDS row stride 40 halves.
// D# encoding per CDNA5 ISA (6-arg clang-23 builtin form).
__device__ __forceinline__ void tdm_load_tile(const _Float16* gsrc, unsigned lds_off,
                                              int rows, int K) {
    unsigned long long ga = (unsigned long long)(uintptr_t)gsrc;
    v4u g0 = { 1u,                                   // count=1, user mode
               lds_off,                              // lds_addr
               (unsigned)(ga & 0xFFFFFFFFu),         // global_addr[31:0]
               (unsigned)((ga >> 32) & 0x1FFFFFFu) | (2u << 30) }; // ga[56:32] | type=2
    v8i g1 = { (1 << 16) | (1 << 20) | (3 << 22) | (3 << 25),
               // data_size=2B | pad_enable | pad_interval=16DW | pad_amount=4DW
               (32 << 16),          // tensor_dim0 = 32  (bits 79:48)
               (rows << 16),        // tensor_dim1 = rows (bits 111:80)
               (32 << 16),          // tile_dim0 = 32    (bits 127:112)
               rows,                // tile_dim1 = rows, tile_dim2 = 0
               K,                   // tensor_dim0_stride[31:0] (elements)
               0, 0 };              // stride hi, tensor_dim1_stride = 0
    v4i z4 = { 0, 0, 0, 0 };
    v8i z8 = { 0, 0, 0, 0, 0, 0, 0, 0 };
    __builtin_amdgcn_tensor_load_to_lds(g0, g1, z4, z4, z8, 0);
}
#else
#define HAVE_TDM 0
#endif

// ---------------------------------------------------------------------------
// f32 -> f16 elementwise convert (n % 1024 == 0)
// ---------------------------------------------------------------------------
__global__ __launch_bounds__(256)
void cvt_f16_kernel(const float* __restrict__ X, _Float16* __restrict__ Y) {
    size_t i = ((size_t)blockIdx.x * 256 + threadIdx.x) * 4;
    float4 f = *(const float4*)(X + i);
    Y[i + 0] = (_Float16)f.x;
    Y[i + 1] = (_Float16)f.y;
    Y[i + 2] = (_Float16)f.z;
    Y[i + 3] = (_Float16)f.w;
}

// ---------------------------------------------------------------------------
// W[K,N] f32 -> Wt[N,K] f16 transpose (32x32 tiles via LDS)
// ---------------------------------------------------------------------------
__global__ __launch_bounds__(256)
void transpose_f16_kernel(const float* __restrict__ W, _Float16* __restrict__ Wt,
                          int Kd, int Nd) {
    __shared__ float T[32][33];
    const int n0 = blockIdx.x * 32, k0 = blockIdx.y * 32;
    const int tx = threadIdx.x & 31, ty = threadIdx.x >> 5;  // ty 0..7
#pragma unroll
    for (int i = 0; i < 4; ++i) {
        int kk = ty * 4 + i;
        T[kk][tx] = W[(size_t)(k0 + kk) * Nd + n0 + tx];
    }
    __syncthreads();
#pragma unroll
    for (int i = 0; i < 4; ++i) {
        int nn = ty * 4 + i;
        Wt[(size_t)(n0 + nn) * Kd + k0 + tx] = (_Float16)T[tx][nn];
    }
}

// ---------------------------------------------------------------------------
// WMMA GEMM: Out[M,N] = act(A[M,K](f16) * Wt[N,K](f16)^T + bias [+ Res])
// Block: 256 threads (8 waves). Block tile 128x64, wave tile 32x32 (2x2 WMMA).
// Tiles DMA'd into LDS by the Tensor Data Mover (double-buffered, TENSORcnt),
// falling back to cooperative v8h copies if the builtin is unavailable.
// M%128==0, N%64==0, K%32==0.
// ---------------------------------------------------------------------------
template <bool OUT_F16, bool RELU, bool RES>
__global__ __launch_bounds__(256)
void gemm_wmma_kernel(const _Float16* __restrict__ A, const _Float16* __restrict__ Wt,
                      const float* __restrict__ bias, const float* __restrict__ Rp,
                      void* __restrict__ Op, int M, int N, int K) {
    __shared__ __align__(16) _Float16 As[2][128 * 40]; // [row][k], stride 40 halves
    __shared__ __align__(16) _Float16 Bs[2][64 * 40];  // [n][k],   stride 40 halves

    const int tid  = threadIdx.x;
    const int lane = tid & 31;
    const int wid  = tid >> 5;
    const int wm   = wid & 3;   // rows 32*wm
    const int wn   = wid >> 2;  // cols 32*wn
    const int bm   = blockIdx.x * 128;
    const int bn   = blockIdx.y * 64;
    const int nl   = lane & 15;
    const int kb   = (lane >> 4) * 8;  // fragment K-base (ISA §7.12.2)
    const int mh   = (lane >> 4) * 8;  // C/D row base

    v8f acc[2][2] = {};
    const int nk = K >> 5;

#if HAVE_TDM
    const unsigned ldsA[2] = { (unsigned)(uintptr_t)&As[0][0], (unsigned)(uintptr_t)&As[1][0] };
    const unsigned ldsB[2] = { (unsigned)(uintptr_t)&Bs[0][0], (unsigned)(uintptr_t)&Bs[1][0] };
    if (wid == 0) {
        tdm_load_tile(A  + (size_t)bm * K, ldsA[0], 128, K);
        tdm_load_tile(Wt + (size_t)bn * K, ldsB[0], 64, K);
    }
    for (int kt = 0; kt < nk; ++kt) {
        const int s = kt & 1;
        if (wid == 0) {
            if (kt + 1 < nk) {   // prefetch next stage into the other buffer
                const int k0n = (kt + 1) << 5;
                tdm_load_tile(A  + (size_t)bm * K + k0n, ldsA[s ^ 1], 128, K);
                tdm_load_tile(Wt + (size_t)bn * K + k0n, ldsB[s ^ 1], 64, K);
                __builtin_amdgcn_s_wait_tensorcnt(2);  // current stage done
            } else {
                __builtin_amdgcn_s_wait_tensorcnt(0);
            }
        }
        __syncthreads();
        fragu af[2], bf[2];
#pragma unroll
        for (int i = 0; i < 2; ++i) {
            const _Float16* p = &As[s][(wm * 32 + i * 16 + nl) * 40 + kb];
            af[i].h[0] = *(const v8h*)p;
            af[i].h[1] = *(const v8h*)(p + 16);
        }
#pragma unroll
        for (int j = 0; j < 2; ++j) {
            const _Float16* p = &Bs[s][(wn * 32 + j * 16 + nl) * 40 + kb];
            bf[j].h[0] = *(const v8h*)p;
            bf[j].h[1] = *(const v8h*)(p + 16);
        }
#pragma unroll
        for (int i = 0; i < 2; ++i)
#pragma unroll
            for (int j = 0; j < 2; ++j)
                acc[i][j] = wmma_f16(af[i].v, bf[j].v, acc[i][j]);
        __syncthreads();  // all waves done with buffer s before it is re-filled
    }
#else
    for (int kt = 0; kt < nk; ++kt) {
        const int k0 = kt << 5;
#pragma unroll
        for (int it = 0; it < 2; ++it) {           // A: 128x32 halves, 512 v8h
            int q = tid + 256 * it, row = q >> 2, col = (q & 3) * 8;
            *(v8h*)&As[0][row * 40 + col] = *(const v8h*)(A + (size_t)(bm + row) * K + k0 + col);
        }
        {                                          // B: 64x32 halves, 256 v8h
            int row = tid >> 2, col = (tid & 3) * 8;
            *(v8h*)&Bs[0][row * 40 + col] = *(const v8h*)(Wt + (size_t)(bn + row) * K + k0 + col);
        }
        __syncthreads();
        fragu af[2], bf[2];
#pragma unroll
        for (int i = 0; i < 2; ++i) {
            const _Float16* p = &As[0][(wm * 32 + i * 16 + nl) * 40 + kb];
            af[i].h[0] = *(const v8h*)p;
            af[i].h[1] = *(const v8h*)(p + 16);
        }
#pragma unroll
        for (int j = 0; j < 2; ++j) {
            const _Float16* p = &Bs[0][(wn * 32 + j * 16 + nl) * 40 + kb];
            bf[j].h[0] = *(const v8h*)p;
            bf[j].h[1] = *(const v8h*)(p + 16);
        }
#pragma unroll
        for (int i = 0; i < 2; ++i)
#pragma unroll
            for (int j = 0; j < 2; ++j)
                acc[i][j] = wmma_f16(af[i].v, bf[j].v, acc[i][j]);
        __syncthreads();
    }
#endif

    const float* Res = Rp;
#pragma unroll
    for (int j = 0; j < 2; ++j) {
        int gn = bn + wn * 32 + j * 16 + nl;
        float bv = bias[gn];
#pragma unroll
        for (int i = 0; i < 2; ++i) {
#pragma unroll
            for (int r = 0; r < 8; ++r) {
                int gm = bm + wm * 32 + i * 16 + mh + r;
                float v = acc[i][j][r] + bv;
                if (RES)  v += Res[(size_t)gm * N + gn];
                if (RELU) v = fmaxf(v, 0.0f);
                if (OUT_F16) ((_Float16*)Op)[(size_t)gm * N + gn] = (_Float16)v;
                else         ((float*)Op)[(size_t)gm * N + gn] = v;
            }
        }
    }
}

// ---------------------------------------------------------------------------
// Banded attention: one wave32 per (b, h, 16-query tile). |i-j| <= 64.
// ---------------------------------------------------------------------------
__global__ __launch_bounds__(32)
void attn_band_kernel(const _Float16* __restrict__ Q, const _Float16* __restrict__ Kp,
                      const _Float16* __restrict__ V, _Float16* __restrict__ AO,
                      int L) {
    const int D = 512;
    const int i0 = blockIdx.x * 16;
    const int h  = blockIdx.y;
    const int b  = blockIdx.z;
    const int lane = threadIdx.x;
    const int nl = lane & 15;
    const int kb = (lane >> 4) * 8;
    const int mrow = (lane >> 4) * 8;

    const size_t base = (size_t)b * L * D + (size_t)h * 64;
    const _Float16* qB = Q  + base;
    const _Float16* kB = Kp + base;
    const _Float16* vB = V  + base;
    _Float16*       oB = AO + base;

    __shared__ __align__(16) _Float16 P[16 * 176];
    __shared__ __align__(16) _Float16 Vs[64 * 40];

    v8h z8 = {};
#pragma unroll
    for (int it = 0; it < 11; ++it) ((v8h*)P)[lane + 32 * it] = z8;

    fragu qf[2];
    {
        const _Float16* qr = qB + (size_t)(i0 + nl) * D;
        qf[0].h[0] = *(const v8h*)(qr + kb);
        qf[0].h[1] = *(const v8h*)(qr + kb + 16);
        qf[1].h[0] = *(const v8h*)(qr + 32 + kb);
        qf[1].h[1] = *(const v8h*)(qr + 32 + kb + 16);
    }

    int jlo = i0 - 64;      if (jlo < 0) jlo = 0;
    int jhi = i0 + 16 + 64; if (jhi > L) jhi = L;
    const int ntiles = (jhi - jlo) >> 4;  // <= 9

    const int MAXT = 9;
    float sc[MAXT][8];
#pragma unroll
    for (int t = 0; t < MAXT; ++t) {
        if (t < ntiles) {
            int j0 = jlo + t * 16;
            const _Float16* kr = kB + (size_t)(j0 + nl) * D;
            fragu kf0, kf1;
            kf0.h[0] = *(const v8h*)(kr + kb);
            kf0.h[1] = *(const v8h*)(kr + kb + 16);
            kf1.h[0] = *(const v8h*)(kr + 32 + kb);
            kf1.h[1] = *(const v8h*)(kr + 32 + kb + 16);
            v8f c = {};
            c = wmma_f16(qf[0].v, kf0.v, c);
            c = wmma_f16(qf[1].v, kf1.v, c);
            int j = j0 + nl;
#pragma unroll
            for (int r = 0; r < 8; ++r) {
                int i = i0 + mrow + r;
                int d = i - j;
                float s = c[r] * 0.125f;  // 1/sqrt(64)
                sc[t][r] = (d <= 64 && d >= -64) ? s : -3.0e38f;
            }
        } else {
#pragma unroll
            for (int r = 0; r < 8; ++r) sc[t][r] = -3.0e38f;
        }
    }

    float rinv[8];
#pragma unroll
    for (int r = 0; r < 8; ++r) {
        float mx = sc[0][r];
#pragma unroll
        for (int t = 1; t < MAXT; ++t) mx = fmaxf(mx, sc[t][r]);
        mx = fmaxf(mx, __shfl_xor(mx, 1, 16));
        mx = fmaxf(mx, __shfl_xor(mx, 2, 16));
        mx = fmaxf(mx, __shfl_xor(mx, 4, 16));
        mx = fmaxf(mx, __shfl_xor(mx, 8, 16));
        float sum = 0.0f;
#pragma unroll
        for (int t = 0; t < MAXT; ++t) {
            if (t < ntiles) {
                float p = __expf(sc[t][r] - mx);
                sum += p;
                P[(mrow + r) * 176 + t * 16 + nl] = (_Float16)p;
            }
        }
        sum += __shfl_xor(sum, 1, 16);
        sum += __shfl_xor(sum, 2, 16);
        sum += __shfl_xor(sum, 4, 16);
        sum += __shfl_xor(sum, 8, 16);
        rinv[r] = 1.0f / sum;
    }
    __syncthreads();

    v8f acc[4] = {};
    const int nchunks = (ntiles + 1) >> 1;  // <= 5
#pragma unroll
    for (int c = 0; c < 5; ++c) {
        if (c < nchunks) {
            __syncthreads();
            {
                int jr = jlo + 32 * c + lane;
                if (jr > L - 1) jr = L - 1;  // P is 0 there
                const _Float16* vr = vB + (size_t)jr * D;
#pragma unroll
                for (int g = 0; g < 8; ++g) {
                    v8h vv = *(const v8h*)(vr + g * 8);
#pragma unroll
                    for (int e = 0; e < 8; ++e) Vs[(g * 8 + e) * 40 + lane] = vv[e];
                }
            }
            __syncthreads();
            fragu pa;
            pa.h[0] = *(const v8h*)(&P[nl * 176 + 32 * c + kb]);
            pa.h[1] = *(const v8h*)(&P[nl * 176 + 32 * c + kb + 16]);
#pragma unroll
            for (int nt = 0; nt < 4; ++nt) {
                fragu vf;
                const _Float16* vrow = &Vs[(nt * 16 + nl) * 40];
                vf.h[0] = *(const v8h*)(vrow + kb);
                vf.h[1] = *(const v8h*)(vrow + kb + 16);
                acc[nt] = wmma_f16(pa.v, vf.v, acc[nt]);
            }
        }
    }

#pragma unroll
    for (int nt = 0; nt < 4; ++nt) {
#pragma unroll
        for (int r = 0; r < 8; ++r) {
            int t_out = i0 + mrow + r;
            oB[(size_t)t_out * D + nt * 16 + nl] = (_Float16)(acc[nt][r] * rinv[r]);
        }
    }
}

// ---------------------------------------------------------------------------
// LayerNorm over D=512, one 256-thread block per row; optional f16 copy.
// ---------------------------------------------------------------------------
__global__ __launch_bounds__(256)
void layernorm_kernel(const float* __restrict__ X, const float* __restrict__ g,
                      const float* __restrict__ bta, float* __restrict__ Out,
                      _Float16* __restrict__ Out16) {
    const int row = blockIdx.x;
    const int tid = threadIdx.x;
    const float* x = X + (size_t)row * 512;
    float v0 = x[tid], v1 = x[tid + 256];
    float s = v0 + v1, ss = v0 * v0 + v1 * v1;
#pragma unroll
    for (int off = 16; off >= 1; off >>= 1) {
        s  += __shfl_xor(s, off, 32);
        ss += __shfl_xor(ss, off, 32);
    }
    __shared__ float rs_[8], rss_[8];
    int lane = tid & 31, wid = tid >> 5;
    if (lane == 0) { rs_[wid] = s; rss_[wid] = ss; }
    __syncthreads();
    float ts = 0.0f, tss = 0.0f;
#pragma unroll
    for (int w = 0; w < 8; ++w) { ts += rs_[w]; tss += rss_[w]; }
    float mu   = ts * (1.0f / 512.0f);
    float var  = tss * (1.0f / 512.0f) - mu * mu;
    float rstd = rsqrtf(var + 1e-5f);
    float y0 = (v0 - mu) * rstd * g[tid]       + bta[tid];
    float y1 = (v1 - mu) * rstd * g[tid + 256] + bta[tid + 256];
    Out[(size_t)row * 512 + tid]       = y0;
    Out[(size_t)row * 512 + tid + 256] = y1;
    if (Out16) {
        Out16[(size_t)row * 512 + tid]       = (_Float16)y0;
        Out16[(size_t)row * 512 + tid + 256] = (_Float16)y1;
    }
}

// ---------------------------------------------------------------------------
extern "C" void kernel_launch(void* const* d_in, const int* in_sizes, int n_in,
                              void* d_out, int out_size, void* d_ws, size_t ws_size,
                              hipStream_t stream) {
    const float* x    = (const float*)d_in[0];
    const float* wq   = (const float*)d_in[1];
    const float* bq   = (const float*)d_in[2];
    const float* wk   = (const float*)d_in[3];
    const float* bk   = (const float*)d_in[4];
    const float* wv   = (const float*)d_in[5];
    const float* bv   = (const float*)d_in[6];
    const float* wo   = (const float*)d_in[7];
    const float* bo   = (const float*)d_in[8];
    const float* ln1g = (const float*)d_in[9];
    const float* ln1b = (const float*)d_in[10];
    const float* w1   = (const float*)d_in[11];
    const float* b1   = (const float*)d_in[12];
    const float* w2   = (const float*)d_in[13];
    const float* b2   = (const float*)d_in[14];
    const float* ln2g = (const float*)d_in[15];
    const float* ln2b = (const float*)d_in[16];
    float* out = (float*)d_out;

    const int D = 512, L = 2048, D4 = 4 * D;
    const int M = in_sizes[0] / D;      // B*L = 8192
    const int Bn = M / L;               // batch

    char* ws = (char*)d_ws;
    size_t off = 0;
    _Float16* x16 = (_Float16*)(ws + off); off += (size_t)M * D * 2;
    _Float16* qb  = (_Float16*)(ws + off); off += (size_t)M * D * 2;
    _Float16* kb  = (_Float16*)(ws + off); off += (size_t)M * D * 2;
    _Float16* vb  = (_Float16*)(ws + off); off += (size_t)M * D * 2;
    _Float16* ff1 = (_Float16*)(ws + off); off += (size_t)M * D4 * 2;
    float*    res = (float*)(ws + off);    off += (size_t)M * D * 4;
    float*    hb  = (float*)(ws + off);    off += (size_t)M * D * 4;
    _Float16* h16 = (_Float16*)(ws + off); off += (size_t)M * D * 2;
    _Float16* wqt = (_Float16*)(ws + off); off += (size_t)D * D * 2;
    _Float16* wkt = (_Float16*)(ws + off); off += (size_t)D * D * 2;
    _Float16* wvt = (_Float16*)(ws + off); off += (size_t)D * D * 2;
    _Float16* wot = (_Float16*)(ws + off); off += (size_t)D * D * 2;
    _Float16* w1t = (_Float16*)(ws + off); off += (size_t)D4 * D * 2;  // [2048,512]
    _Float16* w2t = (_Float16*)(ws + off); off += (size_t)D * D4 * 2;  // [512,2048]

    dim3 blk(256);

    // 0) precision prep: x -> f16, weights -> transposed f16 [N,K]
    cvt_f16_kernel<<<dim3((M * D) / 1024), blk, 0, stream>>>(x, x16);
    transpose_f16_kernel<<<dim3(D / 32, D / 32),  blk, 0, stream>>>(wq, wqt, D, D);
    transpose_f16_kernel<<<dim3(D / 32, D / 32),  blk, 0, stream>>>(wk, wkt, D, D);
    transpose_f16_kernel<<<dim3(D / 32, D / 32),  blk, 0, stream>>>(wv, wvt, D, D);
    transpose_f16_kernel<<<dim3(D / 32, D / 32),  blk, 0, stream>>>(wo, wot, D, D);
    transpose_f16_kernel<<<dim3(D4 / 32, D / 32), blk, 0, stream>>>(w1, w1t, D, D4);
    transpose_f16_kernel<<<dim3(D / 32, D4 / 32), blk, 0, stream>>>(w2, w2t, D4, D);

    dim3 g512(M / 128, D / 64);
    dim3 g2048(M / 128, D4 / 64);

    // 1) QKV projections
    gemm_wmma_kernel<true, false, false><<<g512, blk, 0, stream>>>(x16, wqt, bq, nullptr, qb, M, D, D);
    gemm_wmma_kernel<true, false, false><<<g512, blk, 0, stream>>>(x16, wkt, bk, nullptr, kb, M, D, D);
    gemm_wmma_kernel<true, false, false><<<g512, blk, 0, stream>>>(x16, wvt, bv, nullptr, vb, M, D, D);

    // 2) banded attention; output overwrites qb
    attn_band_kernel<<<dim3(L / 16, 8, Bn), dim3(32), 0, stream>>>(qb, kb, vb, qb, L);

    // 3) output projection + residual(x) -> res (f32)
    gemm_wmma_kernel<false, false, true><<<g512, blk, 0, stream>>>(qb, wot, bo, x, res, M, D, D);

    // 4) layernorm1 -> hb (f32) + h16 (f16)
    layernorm_kernel<<<dim3(M), blk, 0, stream>>>(res, ln1g, ln1b, hb, h16);

    // 5) FFN up + ReLU -> ff1 (f16)
    gemm_wmma_kernel<true, true, false><<<g2048, blk, 0, stream>>>(h16, w1t, b1, nullptr, ff1, M, D4, D);

    // 6) FFN down + residual(hb) -> res (f32, reuse)
    gemm_wmma_kernel<false, false, true><<<g512, blk, 0, stream>>>(ff1, w2t, b2, hb, res, M, D, D4);

    // 7) layernorm2 -> d_out
    layernorm_kernel<<<dim3(M), blk, 0, stream>>>(res, ln2g, ln2b, out, nullptr);
}
